// GNN_54322746360586
// MI455X (gfx1250) — matrix operand order repaired and checked by reading, not compiled, not run on hardware
//
#include <hip/hip_runtime.h>
#include <cstdint>

#define NNODES 50000
#define NEDGES 800000
#define NGRAPH 500
#define NTILES 3125   /* NNODES / 16 exactly */
#define BN_EPS 1e-5f

typedef __attribute__((ext_vector_type(16))) __bf16        v16bf;
typedef __attribute__((ext_vector_type(8)))  float         v8f;
typedef __attribute__((ext_vector_type(4)))  unsigned int  v4u;

union ABReg { v4u u[2]; v16bf bf; };

__device__ __forceinline__ unsigned short f2bf(float f) {
  unsigned int u = __float_as_uint(f);
  u += 0x7fffu + ((u >> 16) & 1u);          // round-to-nearest-even
  return (unsigned short)(u >> 16);
}
__device__ __forceinline__ unsigned int fkey(float f) {    // order-preserving float->uint
  unsigned int u = __float_as_uint(f);
  return (u & 0x80000000u) ? ~u : (u | 0x80000000u);
}
__device__ __forceinline__ float funkey(unsigned int k) {
  unsigned int u = (k & 0x80000000u) ? (k ^ 0x80000000u) : ~k;
  return __uint_as_float(u);
}
__device__ __forceinline__ bool finitef(float f) {
  return (__float_as_uint(f) & 0x7f800000u) != 0x7f800000u;
}
__device__ __forceinline__ float lrelu(float x) { return x > 0.f ? x : 0.2f * x; }
__device__ __forceinline__ float wave_sum(float s) {
#pragma unroll
  for (int o = 16; o > 0; o >>= 1) s += __shfl_xor(s, o, 32);
  return s;
}

// ---------------- utility kernels ----------------
__global__ void fill_f32_kernel(float* p, float v, size_t n) {
  size_t i = (size_t)blockIdx.x * blockDim.x + threadIdx.x;
  if (i < n) p[i] = v;
}
__global__ void fill_u32_kernel(unsigned int* p, unsigned int v, size_t n) {
  size_t i = (size_t)blockIdx.x * blockDim.x + threadIdx.x;
  if (i < n) p[i] = v;
}
// pack two f32 -> one dword of two bf16 (n is element count, even)
__global__ void cvt_bf16_kernel(const float* __restrict__ src,
                                unsigned int* __restrict__ dst, size_t npairs) {
  size_t i = (size_t)blockIdx.x * blockDim.x + threadIdx.x;
  if (i >= npairs) return;
  unsigned int lo = f2bf(src[2 * i]);
  unsigned int hi = f2bf(src[2 * i + 1]);
  dst[i] = lo | (hi << 16);
}

// Repack W [K,M] row-major f32 -> packed bf16 pairs for WMMA B fragments.
// Wpk[((ct*nkt + kt)*32 + lane)*8 + v] = {W[k0,n], W[k0+1,n]} with
//   k0 = kt*32 + 16*(lane>>4) + 2*v, n = ct*16 + (lane&15)
__global__ void repack_W_kernel(const float* __restrict__ W,
                                unsigned int* __restrict__ Wpk, int K, int M) {
  int t = blockIdx.x * blockDim.x + threadIdx.x;
  int total = (K * M) >> 1;
  if (t >= total) return;
  int nkt = K >> 5;
  int v    = t & 7;
  int lane = (t >> 3) & 31;
  int kt   = (t >> 8) % nkt;
  int ct   = (t >> 8) / nkt;
  int k0 = kt * 32 + 16 * (lane >> 4) + 2 * v;
  int n  = ct * 16 + (lane & 15);
  unsigned int lo = f2bf(W[(size_t)k0 * M + n]);
  unsigned int hi = f2bf(W[(size_t)(k0 + 1) * M + n]);
  Wpk[t] = lo | (hi << 16);
}

// H[N,M] = XB[N,K](bf16) @ W(K,M packed bf16), f32 accumulate via WMMA.
// One wave per (node_tile, col_tile); K in steps of 32.
__global__ __launch_bounds__(256)
void wmma_gemm_kernel(const unsigned short* __restrict__ XB,
                      const unsigned int* __restrict__ Wpk,
                      float* __restrict__ Hout, int K, int M) {
  const int lane = threadIdx.x & 31;
  const int wid  = threadIdx.x >> 5;
  const int nct  = M >> 4;
  const int w    = blockIdx.x * 8 + wid;    // grid sized exactly: no tail
  const int nt   = w / nct;
  const int ct   = w % nct;
  const int nkt  = K >> 5;
  const int half = lane >> 4;
  const int row  = nt * 16 + (lane & 15);

  const unsigned int* arow  = (const unsigned int*)XB + (size_t)row * (K >> 1);
  const unsigned int* bbase = Wpk + ((size_t)ct * nkt * 32 + lane) * 8;

  v8f acc = {};
  for (int kt = 0; kt < nkt; ++kt) {
    ABReg a, b;
    const unsigned int* ap = arow + kt * 16 + half * 8;
    a.u[0] = *(const v4u*)(ap);        // VGPRs 0..3 : K = 0..7  (+16 for upper half-lanes)
    a.u[1] = *(const v4u*)(ap + 4);    // VGPRs 4..7 : K = 8..15 (+16)
    const unsigned int* bp = bbase + (size_t)kt * 256;
    b.u[0] = *(const v4u*)(bp);
    b.u[1] = *(const v4u*)(bp + 4);
    acc = __builtin_amdgcn_wmma_f32_16x16x32_bf16(
        false, a.bf, false, b.bf, (short)0, acc, false, false);
  }
  // D layout: element (m = v + 8*half, n = lane&15)
  float* o = Hout + (size_t)(nt * 16) * M + ct * 16 + (lane & 15);
#pragma unroll
  for (int v = 0; v < 8; ++v) o[(size_t)(v + 8 * half) * M] = acc[v];
}

// a_src[n,h] = sum_c h[n,h,c]*att_src[h,c]; wave per (n,h), coalesced reads
__global__ void att_scores_kernel(const float* __restrict__ Hf,
                                  const float* __restrict__ asp,
                                  const float* __restrict__ adp,
                                  float* __restrict__ a_src,
                                  float* __restrict__ a_dst, int C) {
  int t = blockIdx.x * blockDim.x + threadIdx.x;
  int w = t >> 5, lane = t & 31;
  if (w >= NNODES * 8) return;
  int n = w >> 3, h = w & 7;
  const float* hr = Hf + (size_t)n * (8 * C) + h * C;
  const float* sa = asp + h * C;
  const float* da = adp + h * C;
  float ss = 0.f, sd = 0.f;
  for (int c = lane; c < C; c += 32) { float v = hr[c]; ss += v * sa[c]; sd += v * da[c]; }
  ss = wave_sum(ss); sd = wave_sum(sd);
  if (lane == 0) { a_src[w] = ss; a_dst[w] = sd; }
}

__global__ void edge_max_kernel(const int* __restrict__ ei,
                                const float* __restrict__ a_src,
                                const float* __restrict__ a_dst,
                                unsigned int* __restrict__ mu) {
  int t = blockIdx.x * blockDim.x + threadIdx.x;
  if (t >= NEDGES * 8) return;
  int e = t >> 3, h = t & 7;
  int s = ei[e], d = ei[NEDGES + e];
  float el = lrelu(a_src[s * 8 + h] + a_dst[d * 8 + h]);
  atomicMax(&mu[d * 8 + h], fkey(el));
}

__global__ void fixup_max_kernel(const unsigned int* __restrict__ mu,
                                 float* __restrict__ mf, size_t n) {
  size_t i = (size_t)blockIdx.x * blockDim.x + threadIdx.x;
  if (i >= n) return;
  float f = funkey(mu[i]);
  mf[i] = finitef(f) ? f : 0.f;
}

__global__ void edge_expsum_kernel(const int* __restrict__ ei,
                                   const float* __restrict__ a_src,
                                   const float* __restrict__ a_dst,
                                   const float* __restrict__ mf,
                                   float* __restrict__ den) {
  int t = blockIdx.x * blockDim.x + threadIdx.x;
  if (t >= NEDGES * 8) return;
  int e = t >> 3, h = t & 7;
  int s = ei[e], d = ei[NEDGES + e];
  float el = lrelu(a_src[s * 8 + h] + a_dst[d * 8 + h]);
  atomicAdd(&den[d * 8 + h], __expf(el - mf[d * 8 + h]));
}

__global__ void init_bias_kernel(float* __restrict__ o,
                                 const float* __restrict__ bias, int Mmask, size_t n) {
  size_t i = (size_t)blockIdx.x * blockDim.x + threadIdx.x;
  if (i < n) o[i] = bias[i & Mmask];
}

// Thread per (edge, head, channel): wave-uniform metadata loads, coalesced
// gather of h[src] and coalesced same-cacheline atomic adds to out[dst].
__global__ void edge_scatter_kernel(const int* __restrict__ ei,
                                    const float* __restrict__ Hf,
                                    const float* __restrict__ a_src,
                                    const float* __restrict__ a_dst,
                                    const float* __restrict__ mf,
                                    const float* __restrict__ den,
                                    float* __restrict__ out, int C, int logC) {
  int t = blockIdx.x * blockDim.x + threadIdx.x;
  if (t >= (NEDGES * 8) << logC) return;
  int c  = t & (C - 1);
  int eh = t >> logC;
  int e = eh >> 3, h = eh & 7;
  int s = ei[e], d = ei[NEDGES + e];
  float el = lrelu(a_src[s * 8 + h] + a_dst[d * 8 + h]);
  float ex = __expf(el - mf[d * 8 + h]);
  float alpha = ex / (den[d * 8 + h] + 1e-16f);
  float v = Hf[(size_t)s * (8 * C) + h * C + c] * alpha;
  atomicAdd(&out[(size_t)d * (8 * C) + h * C + c], v);
}

// Coalesced BN partial sums: threadIdx.x = feature within 256-col group,
// blockIdx.x tiles rows, blockIdx.y selects column group.
__global__ void bn_partial_kernel(const float* __restrict__ X,
                                  float* __restrict__ sum,
                                  float* __restrict__ sumsq,
                                  int M, int rows_per_blk) {
  int f  = blockIdx.y * 256 + threadIdx.x;
  int n0 = blockIdx.x * rows_per_blk;
  int n1 = n0 + rows_per_blk; if (n1 > NNODES) n1 = NNODES;
  float a = 0.f, b = 0.f;
  for (int n = n0; n < n1; ++n) {
    float v = X[(size_t)n * M + f];
    a += v; b += v * v;
  }
  atomicAdd(&sum[f], a);
  atomicAdd(&sumsq[f], b);
}
__global__ void bn_finalize_kernel(float* __restrict__ sum,    // -> mean
                                   float* __restrict__ sumsq,  // -> var
                                   int M) {
  int f = blockIdx.x * blockDim.x + threadIdx.x;
  if (f >= M) return;
  float mu = sum[f] / (float)NNODES;
  sum[f]   = mu;
  sumsq[f] = sumsq[f] / (float)NNODES - mu * mu;
}

__global__ void bn_relu_kernel(float* __restrict__ X,
                               const float* __restrict__ mean,
                               const float* __restrict__ var,
                               const float* __restrict__ g,
                               const float* __restrict__ be, int Mmask, size_t n) {
  size_t i = (size_t)blockIdx.x * blockDim.x + threadIdx.x;
  if (i >= n) return;
  int f = (int)(i & Mmask);
  float v = (X[i] - mean[f]) * rsqrtf(var[f] + BN_EPS) * g[f] + be[f];
  X[i] = fmaxf(v, 0.f);
}

// wave per node: coalesced dot(h[n,:], gw)
__global__ void gate_kernel(const float* __restrict__ Hf,
                            const float* __restrict__ gw,
                            const float* __restrict__ gb,
                            float* __restrict__ gate) {
  int t = blockIdx.x * blockDim.x + threadIdx.x;
  int n = t >> 5, lane = t & 31;
  if (n >= NNODES) return;
  const float* hr = Hf + (size_t)n * 256;
  float s = 0.f;
  for (int c = lane; c < 256; c += 32) s += hr[c] * gw[c];
  s = wave_sum(s);
  if (lane == 0) gate[n] = s + gb[0];
}

__global__ void graph_max_kernel(const int* __restrict__ batch,
                                 const float* __restrict__ gate,
                                 unsigned int* __restrict__ gmu) {
  int n = blockIdx.x * blockDim.x + threadIdx.x;
  if (n >= NNODES) return;
  atomicMax(&gmu[batch[n]], fkey(gate[n]));
}

__global__ void graph_expsum_kernel(const int* __restrict__ batch,
                                    const float* __restrict__ gate,
                                    const float* __restrict__ gmf,
                                    float* __restrict__ gden) {
  int n = blockIdx.x * blockDim.x + threadIdx.x;
  if (n >= NNODES) return;
  int g = batch[n];
  atomicAdd(&gden[g], __expf(gate[n] - gmf[g]));
}

// thread per (node, channel): coalesced reads + coalesced atomics
__global__ void pool_scatter_kernel(const int* __restrict__ batch,
                                    const float* __restrict__ gate,
                                    const float* __restrict__ gmf,
                                    const float* __restrict__ gden,
                                    const float* __restrict__ Hf,
                                    float* __restrict__ pooled) {
  int t = blockIdx.x * blockDim.x + threadIdx.x;
  if (t >= NNODES * 256) return;
  int c = t & 255, n = t >> 8;
  int g = batch[n];
  float w = __expf(gate[n] - gmf[g]) / (gden[g] + 1e-16f);
  atomicAdd(&pooled[(size_t)g * 256 + c], Hf[(size_t)n * 256 + c] * w);
}

__global__ void fc1_kernel(const float* __restrict__ pooled,
                           const float* __restrict__ f1w,
                           const float* __restrict__ f1b,
                           float* __restrict__ z) {
  int t = blockIdx.x * blockDim.x + threadIdx.x;
  if (t >= NGRAPH * 16) return;
  int g = t / 16, j = t % 16;
  const float* pr = pooled + (size_t)g * 256;
  float s = 0.f;
  for (int c = 0; c < 256; ++c) s += pr[c] * f1w[c * 16 + j];
  z[t] = fmaxf(s + f1b[j], 0.f);
}

__global__ void head_kernel(const float* __restrict__ z,
                            const float* __restrict__ ow,
                            const float* __restrict__ ob,
                            float* __restrict__ q) {
  int t = blockIdx.x * blockDim.x + threadIdx.x;
  if (t >= NGRAPH * 200) return;
  int g = t / 200, o = t % 200;
  const float* zr = z + (size_t)g * 16;
  float s = 0.f;
  for (int j = 0; j < 16; ++j) s += zr[j] * ow[j * 200 + o];
  q[t] = s + ob[o];
}

// ---------------- host orchestration ----------------
static inline unsigned int blks(size_t n) { return (unsigned int)((n + 255) / 256); }
static inline int ilog2(int v) { int l = 0; while ((1 << l) < v) ++l; return l; }

static void run_gat_layer(hipStream_t stream, const float* Xin, int K, int M, int C,
                          const float* W, const float* asp, const float* adp,
                          const float* bias, const float* gamma, const float* beta,
                          const int* ei,
                          unsigned short* XB, unsigned int* Wpk,
                          float* F1, float* F2,
                          float* a_src, float* a_dst, unsigned int* mu,
                          float* mf, float* den, float* bnm, float* bnv) {
  size_t nx = (size_t)NNODES * K;
  cvt_bf16_kernel<<<blks(nx / 2), 256, 0, stream>>>(Xin, (unsigned int*)XB, nx / 2);
  int npk = (K * M) >> 1;
  repack_W_kernel<<<blks((size_t)npk), 256, 0, stream>>>(W, Wpk, K, M);
  int nct = M >> 4;
  wmma_gemm_kernel<<<(NTILES * nct) / 8, 256, 0, stream>>>(XB, Wpk, F1, K, M);
  att_scores_kernel<<<blks((size_t)NNODES * 8 * 32), 256, 0, stream>>>(F1, asp, adp, a_src, a_dst, C);
  fill_u32_kernel<<<blks((size_t)NNODES * 8), 256, 0, stream>>>(mu, 0u, (size_t)NNODES * 8);
  edge_max_kernel<<<blks((size_t)NEDGES * 8), 256, 0, stream>>>(ei, a_src, a_dst, mu);
  fixup_max_kernel<<<blks((size_t)NNODES * 8), 256, 0, stream>>>(mu, mf, (size_t)NNODES * 8);
  fill_f32_kernel<<<blks((size_t)NNODES * 8), 256, 0, stream>>>(den, 0.f, (size_t)NNODES * 8);
  edge_expsum_kernel<<<blks((size_t)NEDGES * 8), 256, 0, stream>>>(ei, a_src, a_dst, mf, den);
  size_t nout = (size_t)NNODES * M;
  init_bias_kernel<<<blks(nout), 256, 0, stream>>>(F2, bias, M - 1, nout);
  edge_scatter_kernel<<<blks((size_t)NEDGES * 8 * C), 256, 0, stream>>>(
      ei, F1, a_src, a_dst, mf, den, F2, C, ilog2(C));
  fill_f32_kernel<<<blks((size_t)M), 256, 0, stream>>>(bnm, 0.f, (size_t)M);
  fill_f32_kernel<<<blks((size_t)M), 256, 0, stream>>>(bnv, 0.f, (size_t)M);
  {
    const int rows_per_blk = 250;   // 200 row chunks
    dim3 grid((NNODES + rows_per_blk - 1) / rows_per_blk, M / 256);
    bn_partial_kernel<<<grid, 256, 0, stream>>>(F2, bnm, bnv, M, rows_per_blk);
  }
  bn_finalize_kernel<<<blks((size_t)M), 256, 0, stream>>>(bnm, bnv, M);
  bn_relu_kernel<<<blks(nout), 256, 0, stream>>>(F2, bnm, bnv, gamma, beta, M - 1, nout);
}

extern "C" void kernel_launch(void* const* d_in, const int* in_sizes, int n_in,
                              void* d_out, int out_size, void* d_ws, size_t ws_size,
                              hipStream_t stream) {
  (void)in_sizes; (void)n_in; (void)out_size; (void)ws_size;
  const float* x     = (const float*)d_in[0];
  const int*   ei    = (const int*)d_in[1];
  const int*   batch = (const int*)d_in[2];
  // d_in[3] = num_assets (scalar, output dims fixed by param shapes)
  const float* W1 = (const float*)d_in[4];  const float* as1 = (const float*)d_in[5];
  const float* ad1= (const float*)d_in[6];  const float* b1  = (const float*)d_in[7];
  const float* g1 = (const float*)d_in[8];  const float* be1 = (const float*)d_in[9];
  const float* W2 = (const float*)d_in[10]; const float* as2 = (const float*)d_in[11];
  const float* ad2= (const float*)d_in[12]; const float* b2  = (const float*)d_in[13];
  const float* g2 = (const float*)d_in[14]; const float* be2 = (const float*)d_in[15];
  const float* W3 = (const float*)d_in[16]; const float* as3 = (const float*)d_in[17];
  const float* ad3= (const float*)d_in[18]; const float* b3  = (const float*)d_in[19];
  const float* g3 = (const float*)d_in[20]; const float* be3 = (const float*)d_in[21];
  const float* gw = (const float*)d_in[22]; const float* gb  = (const float*)d_in[23];
  const float* f1w= (const float*)d_in[24]; const float* f1b = (const float*)d_in[25];
  const float* ow = (const float*)d_in[26]; const float* ob  = (const float*)d_in[27];

  char* p = (char*)d_ws;
  auto carve = [&](size_t bytes) -> char* {
    char* r = p; p += (bytes + 255) & ~(size_t)255; return r;
  };
  float*          F1    = (float*)carve((size_t)NNODES * 512 * 4);
  float*          F2    = (float*)carve((size_t)NNODES * 512 * 4);
  unsigned short* XB    = (unsigned short*)carve((size_t)NNODES * 512 * 2);
  unsigned int*   Wpk   = (unsigned int*)carve((size_t)512 * 512 * 2);
  float*          a_src = (float*)carve((size_t)NNODES * 8 * 4);
  float*          a_dst = (float*)carve((size_t)NNODES * 8 * 4);
  unsigned int*   mu    = (unsigned int*)carve((size_t)NNODES * 8 * 4);
  float*          mf    = (float*)carve((size_t)NNODES * 8 * 4);
  float*          den   = (float*)carve((size_t)NNODES * 8 * 4);
  float*          bnm   = (float*)carve(512 * 4);
  float*          bnv   = (float*)carve(512 * 4);
  float*          gate  = (float*)carve((size_t)NNODES * 4);
  unsigned int*   gmu   = (unsigned int*)carve(NGRAPH * 4);
  float*          gmf   = (float*)carve(NGRAPH * 4);
  float*          gden  = (float*)carve(NGRAPH * 4);
  float*          pooled= (float*)carve((size_t)NGRAPH * 256 * 4);
  float*          z     = (float*)carve((size_t)NGRAPH * 16 * 4);

  // conv1: 32 -> 8x32 ; conv2: 256 -> 8x64 ; conv3: 512 -> 8x32
  run_gat_layer(stream, x,  32,  256, 32, W1, as1, ad1, b1, g1, be1, ei,
                XB, Wpk, F1, F2, a_src, a_dst, mu, mf, den, bnm, bnv);
  run_gat_layer(stream, F2, 256, 512, 64, W2, as2, ad2, b2, g2, be2, ei,
                XB, Wpk, F1, F2, a_src, a_dst, mu, mf, den, bnm, bnv);
  run_gat_layer(stream, F2, 512, 256, 32, W3, as3, ad3, b3, g3, be3, ei,
                XB, Wpk, F1, F2, a_src, a_dst, mu, mf, den, bnm, bnv);

  // attentional pooling over graphs
  gate_kernel<<<blks((size_t)NNODES * 32), 256, 0, stream>>>(F2, gw, gb, gate);
  fill_u32_kernel<<<blks(NGRAPH), 256, 0, stream>>>(gmu, 0u, NGRAPH);
  graph_max_kernel<<<blks(NNODES), 256, 0, stream>>>(batch, gate, gmu);
  fixup_max_kernel<<<blks(NGRAPH), 256, 0, stream>>>(gmu, gmf, NGRAPH);
  fill_f32_kernel<<<blks(NGRAPH), 256, 0, stream>>>(gden, 0.f, NGRAPH);
  graph_expsum_kernel<<<blks(NNODES), 256, 0, stream>>>(batch, gate, gmf, gden);
  fill_f32_kernel<<<blks((size_t)NGRAPH * 256), 256, 0, stream>>>(pooled, 0.f, (size_t)NGRAPH * 256);
  pool_scatter_kernel<<<blks((size_t)NNODES * 256), 256, 0, stream>>>(batch, gate, gmf, gden, F2, pooled);
  fc1_kernel<<<blks((size_t)NGRAPH * 16), 256, 0, stream>>>(pooled, f1w, f1b, z);
  head_kernel<<<blks((size_t)NGRAPH * 200), 256, 0, stream>>>(z, ow, ob, (float*)d_out);
}
